// MoE_73985106641134
// MI455X (gfx1250) — matrix operand order repaired and checked by reading, not compiled, not run on hardware
//
#include <hip/hip_runtime.h>

typedef __attribute__((ext_vector_type(16))) __bf16 v16bf;
typedef __attribute__((ext_vector_type(8)))  float  v8f;

#define NB 4096   // tokens = B*S
#define DD 1024   // hidden
#define FF 4096   // intermediate
#define EE 8      // experts
#define KK 2      // top-k

static __device__ __forceinline__ v8f vzero8() {
  v8f z = {0.f,0.f,0.f,0.f,0.f,0.f,0.f,0.f};
  return z;
}

// ---- CDNA5 async memory->LDS copy (ASYNCcnt path), 16 bytes per lane ----
static __device__ __forceinline__ unsigned lds_off(const void* p) {
  // flat->LDS: low 32 bits of the address give the LDS offset
  return (unsigned)(unsigned long long)p;
}
static __device__ __forceinline__ void async_copy16(unsigned ldsAddr, const void* g) {
  unsigned long long ga = (unsigned long long)g;
  asm volatile("global_load_async_to_lds_b128 %0, %1, off"
               :: "v"(ldsAddr), "v"(ga) : "memory");
}
static __device__ __forceinline__ void wait_async0() {
  asm volatile("s_wait_asynccnt 0" ::: "memory");
}

// ---------------- init ----------------
__global__ void k_init(int* __restrict__ cnt) {
  if (threadIdx.x < EE) cnt[threadIdx.x] = 0;
}

// ---------------- router: softmax + top-2 + slot lists ----------------
__global__ __launch_bounds__(128) void k_router(const float* __restrict__ x,
                                                const float* __restrict__ wg,
                                                int* __restrict__ cnt,
                                                int* __restrict__ elist,
                                                float* __restrict__ wslot) {
  int t = blockIdx.x * 128 + threadIdx.x;
  if (t >= NB) return;
  float acc[EE];
#pragma unroll
  for (int e = 0; e < EE; ++e) acc[e] = 0.f;
  const float* xr = x + (size_t)t * DD;
  for (int d = 0; d < DD; ++d) {
    float xv = xr[d];
    const float4* w4 = (const float4*)(wg + (size_t)d * EE);
    float4 a = w4[0], b = w4[1];
    acc[0] += xv * a.x; acc[1] += xv * a.y; acc[2] += xv * a.z; acc[3] += xv * a.w;
    acc[4] += xv * b.x; acc[5] += xv * b.y; acc[6] += xv * b.z; acc[7] += xv * b.w;
  }
  float m = acc[0];
#pragma unroll
  for (int e = 1; e < EE; ++e) m = fmaxf(m, acc[e]);
  float p[EE]; float s = 0.f;
#pragma unroll
  for (int e = 0; e < EE; ++e) { p[e] = __expf(acc[e] - m); s += p[e]; }
  float inv = __builtin_amdgcn_rcpf(s);
#pragma unroll
  for (int e = 0; e < EE; ++e) p[e] *= inv;
  int e0 = 0;
#pragma unroll
  for (int e = 1; e < EE; ++e) if (p[e] > p[e0]) e0 = e;
  int e1 = (e0 == 0) ? 1 : 0;
#pragma unroll
  for (int e = 0; e < EE; ++e) if (e != e0 && p[e] > p[e1]) e1 = e;
  float rden = __builtin_amdgcn_rcpf(fmaxf(p[e0] + p[e1], 1e-9f));
  int s0 = t * KK + 0, s1 = t * KK + 1;
  wslot[s0] = p[e0] * rden;
  wslot[s1] = p[e1] * rden;
  int i0 = atomicAdd(&cnt[e0], 1); elist[e0 * NB + i0] = s0;
  int i1 = atomicAdd(&cnt[e1], 1); elist[e1 * NB + i1] = s1;
}

// ---------------- cast x -> bf16 ----------------
__global__ __launch_bounds__(256) void k_cast(const float* __restrict__ x,
                                              __bf16* __restrict__ xb) {
  int g = blockIdx.x * 256 + threadIdx.x;   // over NB*DD/4 elements
  float4 v = ((const float4*)x)[g];
  union { __bf16 b[4]; uint2 u; } U;
  U.b[0] = (__bf16)v.x; U.b[1] = (__bf16)v.y;
  U.b[2] = (__bf16)v.z; U.b[3] = (__bf16)v.w;
  ((uint2*)xb)[g] = U.u;
}

// ---------------- WMMA fragment loaders (ISA 7.12.2 16-bit layouts) ----------------
static __device__ __forceinline__ v16bf load_a_frag(const __bf16* As, int ldA,
                                                    int row16, int lane) {
  int hi = lane >> 4, m = lane & 15;
  union { unsigned int u[8]; v16bf v; } U;
  const __bf16* rp = As + (size_t)(row16 + m) * ldA;
#pragma unroll
  for (int vi = 0; vi < 8; ++vi) {
    int base = 2 * (vi & 3) + 8 * hi + 16 * (vi >> 2);
    U.u[vi] = *(const unsigned int*)(rp + base);
  }
  return U.v;
}
static __device__ __forceinline__ v16bf load_b_frag(const __bf16* Bt, int ldB,
                                                    int col16, int lane) {
  int hi = lane >> 4, n = lane & 15;
  union { uint2 q[4]; v16bf v; } U;
  const __bf16* cp = Bt + (size_t)(col16 + n) * ldB + 8 * hi;
#pragma unroll
  for (int qi = 0; qi < 4; ++qi) U.q[qi] = *(const uint2*)(cp + 4 * (qi & 1) + 16 * (qi >> 1));
  return U.v;
}

#define WMMA_BF16(A, Bm, C) \
  __builtin_amdgcn_wmma_f32_16x16x32_bf16(false, (A), false, (Bm), (short)0, (C), false, false)

// ---------------- GEMM1: gathered x @ w_gate_up[e], fused SwiGLU ----------------
__global__ __launch_bounds__(256) void k_gemm1(const __bf16* __restrict__ xb,
                                               const float* __restrict__ wgu,
                                               const int* __restrict__ cnt,
                                               const int* __restrict__ elist,
                                               __bf16* __restrict__ act) {
  int e = blockIdx.z;
  int count = cnt[e];
  int m0 = blockIdx.y * 128;
  if (m0 >= count) return;
  int f0 = blockIdx.x * 64;

  __shared__ int rows[128];
  __shared__ int slots[128];
  __shared__ __align__(16) __bf16 As[2][128][32];
  __shared__ __align__(16) __bf16 Bg[2][64][36];
  __shared__ __align__(16) __bf16 Bu[2][64][36];

  int tid = threadIdx.x;
  if (tid < 128) {
    int i = m0 + tid;
    int ic = (i < count) ? i : (count - 1);
    int s = elist[e * NB + ic];
    slots[tid] = (i < count) ? s : -1;
    rows[tid]  = s >> 1;            // slot -> token
  }
  __syncthreads();

  int wave = tid >> 5, lane = tid & 31;
  int wm = (wave & 3) * 32;         // 4-way M split
  int wf = (wave >> 2) * 32;        // 2-way F split
  v8f cg[2][2], cu[2][2];
#pragma unroll
  for (int i = 0; i < 2; ++i)
#pragma unroll
    for (int j = 0; j < 2; ++j) { cg[i][j] = vzero8(); cu[i][j] = vzero8(); }

  const float* Wg = wgu + (size_t)e * DD * (2 * FF);
  int bcol = tid & 63, bks = (tid >> 6) * 8;
  int arow = tid >> 1, aseg = (tid & 1) * 16;
  const __bf16* arowp = xb + (size_t)rows[arow] * DD + aseg;

  float bgr[8], bur[8];
  // ---- pipeline prologue: tile 0 ----
  async_copy16(lds_off(&As[0][arow][aseg]),     arowp);
  async_copy16(lds_off(&As[0][arow][aseg + 8]), arowp + 8);
#pragma unroll
  for (int kk = 0; kk < 8; ++kk) {
    size_t ro = (size_t)(bks + kk) * (2 * FF);
    bgr[kk] = Wg[ro + f0 + bcol];
    bur[kk] = Wg[ro + FF + f0 + bcol];
  }
#pragma unroll
  for (int kk = 0; kk < 8; ++kk) {
    Bg[0][bcol][bks + kk] = (__bf16)bgr[kk];
    Bu[0][bcol][bks + kk] = (__bf16)bur[kk];
  }
  wait_async0();
  __syncthreads();

  const int nIter = DD / 32;
  for (int it = 0; it < nIter; ++it) {
    int cur = it & 1, nxt = cur ^ 1;
    bool more = (it + 1) < nIter;
    if (more) {
      const __bf16* ap = arowp + (it + 1) * 32;
      async_copy16(lds_off(&As[nxt][arow][aseg]),     ap);
      async_copy16(lds_off(&As[nxt][arow][aseg + 8]), ap + 8);
#pragma unroll
      for (int kk = 0; kk < 8; ++kk) {
        size_t ro = (size_t)((it + 1) * 32 + bks + kk) * (2 * FF);
        bgr[kk] = Wg[ro + f0 + bcol];
        bur[kk] = Wg[ro + FF + f0 + bcol];
      }
      if (it + 2 < nIter)
        __builtin_prefetch(Wg + (size_t)((it + 2) * 32 + bks) * (2 * FF) + f0 + bcol, 0, 0);
    }

    v16bf a0  = load_a_frag(&As[cur][0][0], 32, wm,      lane);
    v16bf a1  = load_a_frag(&As[cur][0][0], 32, wm + 16, lane);
    v16bf bg0 = load_b_frag(&Bg[cur][0][0], 36, wf,      lane);
    v16bf bg1 = load_b_frag(&Bg[cur][0][0], 36, wf + 16, lane);
    v16bf bu0 = load_b_frag(&Bu[cur][0][0], 36, wf,      lane);
    v16bf bu1 = load_b_frag(&Bu[cur][0][0], 36, wf + 16, lane);

    cg[0][0] = WMMA_BF16(a0, bg0, cg[0][0]);
    cg[0][1] = WMMA_BF16(a0, bg1, cg[0][1]);
    cg[1][0] = WMMA_BF16(a1, bg0, cg[1][0]);
    cg[1][1] = WMMA_BF16(a1, bg1, cg[1][1]);
    cu[0][0] = WMMA_BF16(a0, bu0, cu[0][0]);
    cu[0][1] = WMMA_BF16(a0, bu1, cu[0][1]);
    cu[1][0] = WMMA_BF16(a1, bu0, cu[1][0]);
    cu[1][1] = WMMA_BF16(a1, bu1, cu[1][1]);

    if (more) {
#pragma unroll
      for (int kk = 0; kk < 8; ++kk) {
        Bg[nxt][bcol][bks + kk] = (__bf16)bgr[kk];
        Bu[nxt][bcol][bks + kk] = (__bf16)bur[kk];
      }
    }
    wait_async0();
    __syncthreads();
  }

  // SwiGLU epilogue (fast rcp-based silu), scatter bf16 act rows by slot
  int hi = lane >> 4, ln = lane & 15;
#pragma unroll
  for (int tm = 0; tm < 2; ++tm)
#pragma unroll
    for (int tn = 0; tn < 2; ++tn)
#pragma unroll
      for (int r = 0; r < 8; ++r) {
        int m = wm + tm * 16 + r + 8 * hi;
        int s = slots[m];
        if (s >= 0) {
          float g = cg[tm][tn][r], u = cu[tm][tn][r];
          float a = g * __builtin_amdgcn_rcpf(1.f + __expf(-g)) * u;
          act[(size_t)s * FF + f0 + wf + tn * 16 + ln] = (__bf16)a;
        }
      }
}

// ---------------- GEMM2: act @ w_down[e] ----------------
__global__ __launch_bounds__(256) void k_gemm2(const __bf16* __restrict__ act,
                                               const float* __restrict__ wd,
                                               const int* __restrict__ cnt,
                                               const int* __restrict__ elist,
                                               float* __restrict__ dout) {
  int e = blockIdx.z;
  int count = cnt[e];
  int m0 = blockIdx.y * 128;
  if (m0 >= count) return;
  int d0 = blockIdx.x * 64;

  __shared__ int rows[128];
  __shared__ int slots[128];
  __shared__ __align__(16) __bf16 As[2][128][32];
  __shared__ __align__(16) __bf16 Bs[2][64][36];

  int tid = threadIdx.x;
  if (tid < 128) {
    int i = m0 + tid;
    int ic = (i < count) ? i : (count - 1);
    int s = elist[e * NB + ic];
    slots[tid] = (i < count) ? s : -1;
    rows[tid]  = s;                 // act row == slot
  }
  __syncthreads();

  int wave = tid >> 5, lane = tid & 31;
  int wm = (wave & 3) * 32;
  int wf = (wave >> 2) * 32;
  v8f c[2][2];
#pragma unroll
  for (int i = 0; i < 2; ++i)
#pragma unroll
    for (int j = 0; j < 2; ++j) c[i][j] = vzero8();

  const float* Wd = wd + (size_t)e * FF * DD;
  int bcol = tid & 63, bks = (tid >> 6) * 8;
  int arow = tid >> 1, aseg = (tid & 1) * 16;
  const __bf16* arowp = act + (size_t)rows[arow] * FF + aseg;

  float br[8];
  // ---- pipeline prologue: tile 0 ----
  async_copy16(lds_off(&As[0][arow][aseg]),     arowp);
  async_copy16(lds_off(&As[0][arow][aseg + 8]), arowp + 8);
#pragma unroll
  for (int kk = 0; kk < 8; ++kk)
    br[kk] = Wd[(size_t)(bks + kk) * DD + d0 + bcol];
#pragma unroll
  for (int kk = 0; kk < 8; ++kk)
    Bs[0][bcol][bks + kk] = (__bf16)br[kk];
  wait_async0();
  __syncthreads();

  const int nIter = FF / 32;
  for (int it = 0; it < nIter; ++it) {
    int cur = it & 1, nxt = cur ^ 1;
    bool more = (it + 1) < nIter;
    if (more) {
      const __bf16* ap = arowp + (it + 1) * 32;
      async_copy16(lds_off(&As[nxt][arow][aseg]),     ap);
      async_copy16(lds_off(&As[nxt][arow][aseg + 8]), ap + 8);
#pragma unroll
      for (int kk = 0; kk < 8; ++kk)
        br[kk] = Wd[(size_t)((it + 1) * 32 + bks + kk) * DD + d0 + bcol];
      if (it + 2 < nIter)
        __builtin_prefetch(Wd + (size_t)((it + 2) * 32 + bks) * DD + d0 + bcol, 0, 0);
    }

    v16bf a0 = load_a_frag(&As[cur][0][0], 32, wm,      lane);
    v16bf a1 = load_a_frag(&As[cur][0][0], 32, wm + 16, lane);
    v16bf b0 = load_b_frag(&Bs[cur][0][0], 36, wf,      lane);
    v16bf b1 = load_b_frag(&Bs[cur][0][0], 36, wf + 16, lane);

    c[0][0] = WMMA_BF16(a0, b0, c[0][0]);
    c[0][1] = WMMA_BF16(a0, b1, c[0][1]);
    c[1][0] = WMMA_BF16(a1, b0, c[1][0]);
    c[1][1] = WMMA_BF16(a1, b1, c[1][1]);

    if (more) {
#pragma unroll
      for (int kk = 0; kk < 8; ++kk)
        Bs[nxt][bcol][bks + kk] = (__bf16)br[kk];
    }
    wait_async0();
    __syncthreads();
  }

  int hi = lane >> 4, ln = lane & 15;
#pragma unroll
  for (int tm = 0; tm < 2; ++tm)
#pragma unroll
    for (int tn = 0; tn < 2; ++tn)
#pragma unroll
      for (int r = 0; r < 8; ++r) {
        int m = wm + tm * 16 + r + 8 * hi;
        int s = slots[m];
        if (s >= 0)
          dout[(size_t)s * DD + d0 + wf + tn * 16 + ln] = c[tm][tn][r];
      }
}

// ---------------- combine: out = w0*down(top1) + w1*down(top2) ----------------
__global__ __launch_bounds__(256) void k_combine(const float* __restrict__ dout,
                                                 const float* __restrict__ wslot,
                                                 float* __restrict__ out) {
  int idx = blockIdx.x * 256 + threadIdx.x;   // over NB*DD
  int t = idx >> 10;                          // DD = 1024
  int d = idx & 1023;
  float w0 = wslot[2 * t], w1 = wslot[2 * t + 1];
  out[idx] = w0 * dout[(size_t)(2 * t) * DD + d] +
             w1 * dout[(size_t)(2 * t + 1) * DD + d];
}

extern "C" void kernel_launch(void* const* d_in, const int* in_sizes, int n_in,
                              void* d_out, int out_size, void* d_ws, size_t ws_size,
                              hipStream_t stream) {
  const float* x   = (const float*)d_in[0];   // [NB, DD]
  const float* wg  = (const float*)d_in[1];   // [DD, EE]
  const float* wgu = (const float*)d_in[2];   // [EE, DD, 2*FF]
  const float* wd  = (const float*)d_in[3];   // [EE, FF, DD]
  float* out = (float*)d_out;                 // [NB, DD]

  // workspace layout (~104 MB):
  char* ws = (char*)d_ws;
  int*   cnt   = (int*)ws;                                    // 256 B
  int*   elist = (int*)(ws + 256);                            // EE*NB*4 = 128 KB
  float* wslot = (float*)(ws + 256 + (size_t)EE * NB * 4);    // KK*NB*4 = 32 KB
  char*  p     = ws + 256 + (size_t)EE * NB * 4 + (size_t)KK * NB * 4;
  __bf16* xb   = (__bf16*)p;  p += (size_t)NB * DD * 2;       // 8 MB
  __bf16* act  = (__bf16*)p;  p += (size_t)KK * NB * FF * 2;  // 64 MB
  float*  dout = (float*)p;                                   // 32 MB

  k_init   <<<1, 32, 0, stream>>>(cnt);
  k_router <<<NB / 128, 128, 0, stream>>>(x, wg, cnt, elist, wslot);
  k_cast   <<<(NB * DD / 4) / 256, 256, 0, stream>>>(x, xb);
  k_gemm1  <<<dim3(FF / 64, NB / 128, EE), 256, 0, stream>>>(xb, wgu, cnt, elist, act);
  k_gemm2  <<<dim3(DD / 64, NB / 128, EE), 256, 0, stream>>>(act, wd, cnt, elist, dout);
  k_combine<<<(NB * DD) / 256, 256, 0, stream>>>(dout, wslot, out);
}
